// GVPR3ScorePredictor_16372415333039
// MI455X (gfx1250) — compile-verified, matrix operand order checked by reading, not run on hardware
//
#include <hip/hip_runtime.h>
#include <math.h>

// ---------------- WMMA bf16 plumbing (CDNA5 gfx1250, wave32) ----------------
typedef __attribute__((ext_vector_type(16))) __bf16 v16bf;
typedef __attribute__((ext_vector_type(8)))  float  v8f;

// A-matrix (16xK tile in LDS, row-major, stride ld): per ISA 7.12.2 16-bit A
// layout: lane L holds row M=L%16; element j holds K = k0 + (L<16?0:8) + (j<8?j:j+8).
// j=0..7 and j=8..15 are each contiguous 16-byte runs -> compiler emits ds_load_b128 x2.
__device__ __forceinline__ v16bf frag_a_lds(const __bf16* base, int ld, int k0, int lane) {
  const int r  = lane & 15;
  const int kb = k0 + ((lane & 16) ? 8 : 0);
  v16bf f;
#pragma unroll
  for (int j = 0; j < 16; ++j) {
    const int k = kb + (j < 8 ? j : j + 8);
    f[j] = base[r * ld + k];
  }
  return f;
}

// B-matrix fragment from pre-swizzled weights: per (k-tile, n-tile) each lane's
// 16 bf16 values are contiguous & 32B-aligned -> one v16bf load (global_load_b128 x2).
__device__ __forceinline__ v16bf frag_b_swz(const __bf16* __restrict__ Wf, int ntiles,
                                            int kt, int nt, int lane) {
  const size_t off = (((size_t)kt * ntiles + nt) * 32 + lane) * 16;
  return *(const v16bf*)(Wf + off);
}

__device__ __forceinline__ v8f wmma_bf16(v16bf a, v16bf b, v8f c) {
  // D = A(16x32 bf16) x B(32x16 bf16) + C(16x16 f32)
  return __builtin_amdgcn_wmma_f32_16x16x32_bf16(false, a, false, b, (short)0, c,
                                                 false, false);
}

__device__ __forceinline__ float sigmoidf_(float x) { return 1.f / (1.f + expf(-x)); }

// ------- weight prep: fp32 -> bf16 in fragment-swizzled layout ---------------
// Logical B is [Kpad][N] (K-major; transpose=1 means src is [N][K] i.e. Wh-style).
// Output: Wf[(((kt*(N/16)+nt)*32 + lane)*16 + j] = B[k][n] with
//   n = nt*16 + (lane&15), k = kt*32 + ((lane&16)?8:0) + (j<8?j:j+8).
__global__ void conv_weight_frag(__bf16* __restrict__ dst, const float* __restrict__ src,
                                 int K, int Kpad, int Nn, int transpose) {
  int i = blockIdx.x * blockDim.x + threadIdx.x;
  int total = Kpad * Nn;
  if (i >= total) return;
  int j    = i & 15;
  int lane = (i >> 4) & 31;
  int rem  = i >> 9;                 // tile index
  int ntiles = Nn >> 4;
  int nt = rem % ntiles;
  int kt = rem / ntiles;
  int n = nt * 16 + (lane & 15);
  int k = kt * 32 + ((lane & 16) ? 8 : 0) + (j < 8 ? j : j + 8);
  float v = 0.f;
  if (k < K) v = transpose ? src[n * K + k] : src[k * Nn + n];
  dst[i] = (__bf16)v;
}

// ---------------- generic fused GVP kernel (direct in/out, 16 rows/block) ----
// s_out = [res +] relu([s, |Wh V|] Ws + bs) ; V_out = [res +] (Wv Wh V) * sigmoid(sp Wg + bg)
template <int SI, int VI, int VIP, int H, int SO, int VO, bool RES>
__global__ __launch_bounds__(128) void gvp_kernel(
    const float* __restrict__ s_in, const float* __restrict__ v_in,
    const __bf16* __restrict__ Wh_t,   // swizzled [VIP x H]
    const __bf16* __restrict__ Ws,     // swizzled [(SI+H) x SO]
    const float*  __restrict__ bs,     // [SO]
    const __bf16* __restrict__ Wv_t,   // swizzled [H x VO]
    const __bf16* __restrict__ Wg,     // swizzled [SO x VO]
    const float*  __restrict__ bg,     // [VO]
    float* __restrict__ s_out, float* __restrict__ v_out, int n) {
  constexpr int KS = SI + H;           // multiple of 32 in all instantiations
  __shared__ __bf16 As[16][KS];
  __shared__ __bf16 Av[3][16][VIP];
  __shared__ __bf16 Vh[3][16][H];
  __shared__ float  spf[16][SO];
  __shared__ __bf16 spb[16][SO];
  __shared__ float  gatef[16][VO];

  const int tid = threadIdx.x, lane = tid & 31, wave = tid >> 5;
  const int row0 = blockIdx.x * 16;

  for (int i = tid; i < 16 * SI; i += 128) {
    int m = i / SI, c = i % SI, r = row0 + m;
    As[m][c] = (r < n) ? (__bf16)s_in[(size_t)r * SI + c] : (__bf16)0.f;
  }
  for (int i = tid; i < 16 * VIP; i += 128) {
    int m = i / VIP, c = i % VIP, r = row0 + m;
#pragma unroll
    for (int d = 0; d < 3; ++d) {
      float v = (r < n && c < VI) ? v_in[((size_t)r * VI + c) * 3 + d] : 0.f;
      Av[d][m][c] = (__bf16)v;
    }
  }
  __syncthreads();

  // GEMM1: Vh[d] = Av[d] (16xVIP) x Wh_t (VIPxH)
  for (int t = wave; t < 3 * (H / 16); t += 4) {
    int d = t / (H / 16), ht = t % (H / 16);
    v8f acc = {};
    for (int kt = 0; kt < VIP / 32; ++kt)
      acc = wmma_bf16(frag_a_lds(&Av[d][0][0], VIP, kt * 32, lane),
                      frag_b_swz(Wh_t, H / 16, kt, ht, lane), acc);
    int m0 = (lane & 16) ? 8 : 0, nn = ht * 16 + (lane & 15);
#pragma unroll
    for (int r8 = 0; r8 < 8; ++r8) Vh[d][m0 + r8][nn] = (__bf16)acc[r8];
  }
  __syncthreads();

  // vector norms -> tail of As
  for (int i = tid; i < 16 * H; i += 128) {
    int m = i / H, h = i % H;
    float a = (float)Vh[0][m][h], b = (float)Vh[1][m][h], c = (float)Vh[2][m][h];
    As[m][SI + h] = (__bf16)sqrtf(a * a + b * b + c * c + 1e-8f);
  }
  __syncthreads();

  // GEMM2: sp = relu(As (16xKS) x Ws (KSxSO) + bs)
  for (int t = wave; t < SO / 16; t += 4) {
    v8f acc = {};
    for (int kt = 0; kt < KS / 32; ++kt)
      acc = wmma_bf16(frag_a_lds(&As[0][0], KS, kt * 32, lane),
                      frag_b_swz(Ws, SO / 16, kt, t, lane), acc);
    int m0 = (lane & 16) ? 8 : 0, nn = t * 16 + (lane & 15);
#pragma unroll
    for (int r8 = 0; r8 < 8; ++r8) {
      float v = fmaxf(acc[r8] + bs[nn], 0.f);
      spf[m0 + r8][nn] = v;
      spb[m0 + r8][nn] = (__bf16)v;
    }
  }
  __syncthreads();

  // GEMM3: gate = sigmoid(sp x Wg + bg)
  for (int t = wave; t < VO / 16; t += 4) {
    v8f acc = {};
    for (int kt = 0; kt < SO / 32; ++kt)
      acc = wmma_bf16(frag_a_lds(&spb[0][0], SO, kt * 32, lane),
                      frag_b_swz(Wg, VO / 16, kt, t, lane), acc);
    int m0 = (lane & 16) ? 8 : 0, nn = t * 16 + (lane & 15);
#pragma unroll
    for (int r8 = 0; r8 < 8; ++r8) gatef[m0 + r8][nn] = sigmoidf_(acc[r8] + bg[nn]);
  }
  __syncthreads();

  // GEMM4: Vmu = Vh[d] x Wv_t, then gate + (residual) + store
  for (int t = wave; t < 3 * (VO / 16); t += 4) {
    int d = t / (VO / 16), ot = t % (VO / 16);
    v8f acc = {};
    for (int kt = 0; kt < H / 32; ++kt)
      acc = wmma_bf16(frag_a_lds(&Vh[d][0][0], H, kt * 32, lane),
                      frag_b_swz(Wv_t, VO / 16, kt, ot, lane), acc);
    int m0 = (lane & 16) ? 8 : 0, nn = ot * 16 + (lane & 15);
#pragma unroll
    for (int r8 = 0; r8 < 8; ++r8) {
      int m = m0 + r8, r = row0 + m;
      if (r < n) {
        float val = acc[r8] * gatef[m][nn];
        if constexpr (RES) val += v_in[((size_t)r * VO + nn) * 3 + d];
        v_out[((size_t)r * VO + nn) * 3 + d] = val;
      }
    }
  }
  // scalar output
  for (int i = tid; i < 16 * SO; i += 128) {
    int m = i / SO, c = i % SO, r = row0 + m;
    if (r < n) {
      float v = spf[m][c];
      if constexpr (RES) v += s_in[(size_t)r * SO + c];
      s_out[(size_t)r * SO + c] = v;
    }
  }
}

// ---------------- fused edge-message GVP: gather -> 4 GEMMs -> scatter-add ---
__global__ __launch_bounds__(128) void edge_msg_kernel(
    const float* __restrict__ s,    // [N,128]
    const float* __restrict__ V,    // [N,64,3]
    const float* __restrict__ es,   // [E,64]
    const float* __restrict__ eV,   // [E,32,3]
    const int*   __restrict__ eidx, // [2,E]
    const __bf16* __restrict__ Whm_t,  // swizzled [160 x 160]
    const __bf16* __restrict__ Wsm,    // swizzled [480 x 128]
    const float*  __restrict__ bsm,
    const __bf16* __restrict__ Wvm_t,  // swizzled [160 x 64]
    const __bf16* __restrict__ Wgm,    // swizzled [128 x 64]
    const float*  __restrict__ bgm,
    float* __restrict__ s_acc,  // [N,128] (pre-initialized with s)
    float* __restrict__ V_acc,  // [N,64,3]
    int E) {
  __shared__ __bf16 As[16][480];
  __shared__ __bf16 Av[3][16][160];
  __shared__ __bf16 Vh[3][16][160];
  __shared__ float  spf[16][128];
  __shared__ __bf16 spb[16][128];
  __shared__ float  gatef[16][64];
  __shared__ int    sdst[16];

  const int tid = threadIdx.x, lane = tid & 31, wave = tid >> 5;
  const int e0 = blockIdx.x * 16;
  if (tid < 16) { int e = e0 + tid; sdst[tid] = (e < E) ? eidx[E + e] : -1; }

  // gather scalars: [s[src](128) | s[dst](128) | es(64)]
  for (int i = tid; i < 16 * 320; i += 128) {
    int m = i / 320, c = i % 320, e = e0 + m;
    float v = 0.f;
    if (e < E) {
      int sc = eidx[e], dc = eidx[E + e];
      if (c < 128)      v = s[(size_t)sc * 128 + c];
      else if (c < 256) v = s[(size_t)dc * 128 + (c - 128)];
      else              v = es[(size_t)e * 64 + (c - 256)];
    }
    As[m][c] = (__bf16)v;
  }
  // gather vectors: [V[src](64) | V[dst](64) | eV(32)]
  for (int i = tid; i < 16 * 160; i += 128) {
    int m = i / 160, c = i % 160, e = e0 + m;
    int sc = 0, dc = 0;
    bool ok = e < E;
    if (ok) { sc = eidx[e]; dc = eidx[E + e]; }
#pragma unroll
    for (int d = 0; d < 3; ++d) {
      float v = 0.f;
      if (ok) {
        if (c < 64)       v = V[((size_t)sc * 64 + c) * 3 + d];
        else if (c < 128) v = V[((size_t)dc * 64 + (c - 64)) * 3 + d];
        else              v = eV[((size_t)e * 32 + (c - 128)) * 3 + d];
      }
      Av[d][m][c] = (__bf16)v;
    }
  }
  __syncthreads();

  // GEMM1: Vh[d] = Av[d] (16x160) x Whm_t (160x160) — 30 tiles, 5 k-steps
  for (int t = wave; t < 30; t += 4) {
    int d = t / 10, ht = t % 10;
    v8f acc = {};
    for (int kt = 0; kt < 5; ++kt)
      acc = wmma_bf16(frag_a_lds(&Av[d][0][0], 160, kt * 32, lane),
                      frag_b_swz(Whm_t, 10, kt, ht, lane), acc);
    int m0 = (lane & 16) ? 8 : 0, nn = ht * 16 + (lane & 15);
#pragma unroll
    for (int r8 = 0; r8 < 8; ++r8) Vh[d][m0 + r8][nn] = (__bf16)acc[r8];
  }
  __syncthreads();

  for (int i = tid; i < 16 * 160; i += 128) {
    int m = i / 160, h = i % 160;
    float a = (float)Vh[0][m][h], b = (float)Vh[1][m][h], c = (float)Vh[2][m][h];
    As[m][320 + h] = (__bf16)sqrtf(a * a + b * b + c * c + 1e-8f);
  }
  __syncthreads();

  // GEMM2: sp = relu(As (16x480) x Wsm (480x128) + bsm) — 8 tiles, 15 k-steps
  for (int t = wave; t < 8; t += 4) {
    v8f acc = {};
    for (int kt = 0; kt < 15; ++kt)
      acc = wmma_bf16(frag_a_lds(&As[0][0], 480, kt * 32, lane),
                      frag_b_swz(Wsm, 8, kt, t, lane), acc);
    int m0 = (lane & 16) ? 8 : 0, nn = t * 16 + (lane & 15);
#pragma unroll
    for (int r8 = 0; r8 < 8; ++r8) {
      float v = fmaxf(acc[r8] + bsm[nn], 0.f);
      spf[m0 + r8][nn] = v;
      spb[m0 + r8][nn] = (__bf16)v;
    }
  }
  __syncthreads();

  // GEMM3: gate = sigmoid(sp x Wgm + bgm) — 4 tiles, 4 k-steps
  for (int t = wave; t < 4; t += 4) {
    v8f acc = {};
    for (int kt = 0; kt < 4; ++kt)
      acc = wmma_bf16(frag_a_lds(&spb[0][0], 128, kt * 32, lane),
                      frag_b_swz(Wgm, 4, kt, t, lane), acc);
    int m0 = (lane & 16) ? 8 : 0, nn = t * 16 + (lane & 15);
#pragma unroll
    for (int r8 = 0; r8 < 8; ++r8) gatef[m0 + r8][nn] = sigmoidf_(acc[r8] + bgm[nn]);
  }
  __syncthreads();

  // scatter scalar messages
  for (int i = tid; i < 16 * 128; i += 128) {
    int m = i / 128, c = i % 128, dn = sdst[m];
    if (dn >= 0) atomicAdd(&s_acc[(size_t)dn * 128 + c], spf[m][c]);
  }
  // GEMM4: Vmu = Vh[d] x Wvm_t (160x64); gate + scatter — 12 tiles, 5 k-steps
  for (int t = wave; t < 12; t += 4) {
    int d = t / 4, ot = t % 4;
    v8f acc = {};
    for (int kt = 0; kt < 5; ++kt)
      acc = wmma_bf16(frag_a_lds(&Vh[d][0][0], 160, kt * 32, lane),
                      frag_b_swz(Wvm_t, 4, kt, ot, lane), acc);
    int m0 = (lane & 16) ? 8 : 0, nn = ot * 16 + (lane & 15);
#pragma unroll
    for (int r8 = 0; r8 < 8; ++r8) {
      int m = m0 + r8, dn = sdst[m];
      if (dn >= 0)
        atomicAdd(&V_acc[((size_t)dn * 64 + nn) * 3 + d], acc[r8] * gatef[m][nn]);
    }
  }
}

// ---------------- prediction head (scalar VALU; tiny) ------------------------
__global__ __launch_bounds__(64) void head_kernel(
    const float* __restrict__ s, const float* __restrict__ V,
    const float* __restrict__ Wpred, const float* __restrict__ Wg1,
    const float* __restrict__ bg1, const float* __restrict__ Wg2,
    const float* __restrict__ bg2, float* __restrict__ out, int n) {
  __shared__ float gfeat[192];
  __shared__ float h1[192];
  __shared__ float scale_s;
  const int nid = blockIdx.x, tid = threadIdx.x;
  if (nid >= n) return;
  for (int j = tid; j < 128; j += 64) gfeat[j] = s[(size_t)nid * 128 + j];
  {
    int j = tid;  // 64 threads, 64 vector channels
    const float* vp = &V[((size_t)nid * 64 + j) * 3];
    gfeat[128 + j] = sqrtf(vp[0] * vp[0] + vp[1] * vp[1] + vp[2] * vp[2]);
  }
  __syncthreads();
  for (int j = tid; j < 192; j += 64) {
    float acc = bg1[j];
    for (int k = 0; k < 192; ++k) acc += gfeat[k] * Wg1[k * 192 + j];
    h1[j] = fmaxf(acc, 0.f);
  }
  __syncthreads();
  if (tid == 0) {
    float acc = bg2[0];
    for (int k = 0; k < 192; ++k) acc += h1[k] * Wg2[k];
    scale_s = (acc > 20.f) ? acc : log1pf(expf(acc));
  }
  __syncthreads();
  if (tid < 3) {
    float sc = 0.f;
    for (int v = 0; v < 64; ++v) sc += V[((size_t)nid * 64 + v) * 3 + tid] * Wpred[v];
    out[(size_t)nid * 3 + tid] = sc * scale_s;
  }
}

// ---------------- host orchestration ----------------------------------------
extern "C" void kernel_launch(void* const* d_in, const int* in_sizes, int n_in,
                              void* d_out, int out_size, void* d_ws, size_t ws_size,
                              hipStream_t stream) {
  const float* x    = (const float*)d_in[0];
  const float* vx   = (const float*)d_in[1];
  const int*   eidx = (const int*)d_in[2];
  const float* ea   = (const float*)d_in[3];
  const float* eva  = (const float*)d_in[4];
  // d_in[5] = orientations (unused by the reference computation)
  const float *Whn = (const float*)d_in[6],  *Wsn = (const float*)d_in[7],
              *bsn = (const float*)d_in[8],  *Wvn = (const float*)d_in[9],
              *Wgn = (const float*)d_in[10], *bgn = (const float*)d_in[11];
  const float *Whe = (const float*)d_in[12], *Wse = (const float*)d_in[13],
              *bse = (const float*)d_in[14], *Wve = (const float*)d_in[15],
              *Wge = (const float*)d_in[16], *bge = (const float*)d_in[17];
  const float *Whm = (const float*)d_in[18], *Wsm = (const float*)d_in[19],
              *bsm = (const float*)d_in[20], *Wvm = (const float*)d_in[21],
              *Wgm = (const float*)d_in[22], *bgm = (const float*)d_in[23];
  const float *Whu = (const float*)d_in[24], *Wsu = (const float*)d_in[25],
              *bsu = (const float*)d_in[26], *Wvu = (const float*)d_in[27],
              *Wgu = (const float*)d_in[28], *bgu = (const float*)d_in[29];
  const float *Wpred = (const float*)d_in[30], *Wg1 = (const float*)d_in[31],
              *bg1 = (const float*)d_in[32],   *Wg2 = (const float*)d_in[33],
              *bg2 = (const float*)d_in[34];

  const int N = in_sizes[0] / 64;   // 16000
  const int E = in_sizes[2] / 2;    // 256000

  char* wp = (char*)d_ws;
  auto alloc = [&](size_t bytes) -> void* {
    void* p = (void*)wp;
    wp += (bytes + 255) & ~(size_t)255;
    return p;
  };
  // bf16 fragment-swizzled weights
  __bf16* whn_t = (__bf16*)alloc(32 * 64 * 2);
  __bf16* wsn_b = (__bf16*)alloc(128 * 128 * 2);
  __bf16* wvn_t = (__bf16*)alloc(64 * 64 * 2);
  __bf16* wgn_b = (__bf16*)alloc(128 * 64 * 2);
  __bf16* whe_t = (__bf16*)alloc(32 * 32 * 2);
  __bf16* wse_b = (__bf16*)alloc(64 * 64 * 2);
  __bf16* wve_t = (__bf16*)alloc(32 * 32 * 2);
  __bf16* wge_b = (__bf16*)alloc(64 * 32 * 2);
  __bf16* whm_t = (__bf16*)alloc(160 * 160 * 2);
  __bf16* wsm_b = (__bf16*)alloc(480 * 128 * 2);
  __bf16* wvm_t = (__bf16*)alloc(160 * 64 * 2);
  __bf16* wgm_b = (__bf16*)alloc(128 * 64 * 2);
  __bf16* whu_t = (__bf16*)alloc(64 * 64 * 2);
  __bf16* wsu_b = (__bf16*)alloc(192 * 128 * 2);
  __bf16* wvu_t = (__bf16*)alloc(64 * 64 * 2);
  __bf16* wgu_b = (__bf16*)alloc(128 * 64 * 2);
  // fp32 state
  float* s_a  = (float*)alloc((size_t)N * 128 * 4);
  float* s_b  = (float*)alloc((size_t)N * 128 * 4);
  float* V_a  = (float*)alloc((size_t)N * 192 * 4);
  float* V_b  = (float*)alloc((size_t)N * 192 * 4);
  float* es_b = (float*)alloc((size_t)E * 64 * 4);
  float* eV_b = (float*)alloc((size_t)E * 96 * 4);

  auto conv = [&](__bf16* dst, const float* src, int K, int Kpad, int Nn, int tr) {
    int elems = Kpad * Nn;
    conv_weight_frag<<<(elems + 255) / 256, 256, 0, stream>>>(dst, src, K, Kpad, Nn, tr);
  };
  // vector-mixing weights [out,in] -> transposed K-major; scalar weights already [K,N]
  conv(whn_t, Whn, 4, 32, 64, 1);     conv(wsn_b, Wsn, 128, 128, 128, 0);
  conv(wvn_t, Wvn, 64, 64, 64, 1);    conv(wgn_b, Wgn, 128, 128, 64, 0);
  conv(whe_t, Whe, 1, 32, 32, 1);     conv(wse_b, Wse, 64, 64, 64, 0);
  conv(wve_t, Wve, 32, 32, 32, 1);    conv(wge_b, Wge, 64, 64, 32, 0);
  conv(whm_t, Whm, 160, 160, 160, 1); conv(wsm_b, Wsm, 480, 480, 128, 0);
  conv(wvm_t, Wvm, 160, 160, 64, 1);  conv(wgm_b, Wgm, 128, 128, 64, 0);
  conv(whu_t, Whu, 64, 64, 64, 1);    conv(wsu_b, Wsu, 192, 192, 128, 0);
  conv(wvu_t, Wvu, 64, 64, 64, 1);    conv(wgu_b, Wgu, 128, 128, 64, 0);

  // node embedding: (64s, 4v) -> (128s, 64v), hidden 64
  gvp_kernel<64, 4, 32, 64, 128, 64, false><<<(N + 15) / 16, 128, 0, stream>>>(
      x, vx, whn_t, wsn_b, bsn, wvn_t, wgn_b, bgn, s_a, V_a, N);
  // edge embedding: (32s, 1v) -> (64s, 32v), hidden 32
  gvp_kernel<32, 1, 32, 32, 64, 32, false><<<(E + 15) / 16, 128, 0, stream>>>(
      ea, eva, whe_t, wse_b, bse, wve_t, wge_b, bge, es_b, eV_b, E);

  for (int layer = 0; layer < 3; ++layer) {
    hipMemcpyAsync(s_b, s_a, (size_t)N * 128 * 4, hipMemcpyDeviceToDevice, stream);
    hipMemcpyAsync(V_b, V_a, (size_t)N * 192 * 4, hipMemcpyDeviceToDevice, stream);
    edge_msg_kernel<<<(E + 15) / 16, 128, 0, stream>>>(
        s_a, V_a, es_b, eV_b, eidx, whm_t, wsm_b, bsm, wvm_t, wgm_b, bgm,
        s_b, V_b, E);
    // node update with residual: s_a = s_b + GVP(s_b, V_b)
    gvp_kernel<128, 64, 64, 64, 128, 64, true><<<(N + 15) / 16, 128, 0, stream>>>(
        s_b, V_b, whu_t, wsu_b, bsu, wvu_t, wgu_b, bgu, s_a, V_a, N);
  }

  head_kernel<<<N, 64, 0, stream>>>(s_a, V_a, Wpred, Wg1, bg1, Wg2, bg2,
                                    (float*)d_out, N);
}
